// GNNEdgeClassifier_61787399520742
// MI455X (gfx1250) — matrix operand order repaired and checked by reading, not compiled, not run on hardware
//
#include <hip/hip_runtime.h>

// ---------------------------------------------------------------------------
// GNN edge classifier for MI455X (gfx1250): all GEMMs on v_wmma_f32_16x16x32_bf16
// (bf16 inputs, fp32 accumulate), gather/scatter as coalesced loads + f32 atomics.
// All irregular accesses use uniform-base + u32 element offsets so the compiler
// emits GVS addressing (saddr + 32-bit voffset, scale_offset) instead of
// per-element 64-bit pointer math.
// ---------------------------------------------------------------------------

#define NNODES 50000
#define NEDGES 800000
#define HIDDEN 128
#define EDGEF  64
#define NLAYER 3

typedef __attribute__((ext_vector_type(16))) __bf16 v16bf;
typedef __attribute__((ext_vector_type(8)))  float  v8f;

// fp32 -> bf16 round-to-nearest-even
__device__ __forceinline__ unsigned short f2bf(float f) {
  unsigned int u = __float_as_uint(f);
  u += 0x7FFFu + ((u >> 16) & 1u);
  return (unsigned short)(u >> 16);
}

// --- A-matrix (16x32 MxK, 16-bit) fragment layout (ISA 7.12.2) -------------
// lane = M + 16*khalf, element i: K = 8*khalf + (i&7) + 16*(i>>3)
__device__ __forceinline__ int a_lane(int m, int kl) { return m + (((kl >> 3) & 1) << 4); }
__device__ __forceinline__ int a_elem(int kl)        { return (kl & 7) + ((kl >> 4) << 3); }

// --- B-matrix (32x16 KxN, 16-bit) fragment layout --------------------------
// lane = N + 16*khalf, element i: K = 8*khalf + (i&7) + 16*(i>>3); N = lane&15
__device__ __forceinline__ int b_klocal(int lane, int i) { return ((lane >> 4) << 3) + (i & 7) + ((i >> 3) << 4); }

// ---------------------------------------------------------------------------
// Pack an fp32 weight matrix W[K][N] (row-major) into bf16 B-fragment layout:
// dst[((kc*(N/16)+nt)*32 + lane)*16 + i]
// ---------------------------------------------------------------------------
__global__ void pack_weights(const float* __restrict__ W, unsigned short* __restrict__ dst,
                             int K, int N) {
  int ntiles = N >> 4;
  int total = (K >> 5) * ntiles * 512;
  for (int t = blockIdx.x * blockDim.x + threadIdx.x; t < total;
       t += gridDim.x * blockDim.x) {
    int kc   = t / (ntiles * 512);
    int rem  = t - kc * (ntiles * 512);
    int nt   = rem >> 9;
    int q    = rem & 511;
    int lane = q >> 4;
    int i    = q & 15;
    unsigned k = kc * 32 + b_klocal(lane, i);
    unsigned n = nt * 16 + (lane & 15);
    dst[t] = f2bf(W[k * (unsigned)N + n]);
  }
}

// ---------------------------------------------------------------------------
// Fused GINE message + aggregate:
//   e = edge_attr @ edge_W + eb ; m = relu(x[src] + e) ; aggr[dst] += m
// One wave per 16 edges; 4 waves per block. aggr must be pre-initialized to x.
// ---------------------------------------------------------------------------
__global__ __launch_bounds__(128) void gine_message(
    const float* __restrict__ edge_attr, const int* __restrict__ src,
    const int* __restrict__ dst, const unsigned short* __restrict__ eWf,
    const float* __restrict__ eb, const float* __restrict__ x,
    float* __restrict__ aggr) {
  __shared__ __align__(32) unsigned short afrag[4][2 * 512];  // 2 K-chunks
  __shared__ int sidx[4][16], didx[4][16];
  int tid = threadIdx.x, w = tid >> 5, lane = tid & 31;
  int e0 = (blockIdx.x * 4 + w) * 16;

  if (lane < 16) { sidx[w][lane] = src[e0 + lane]; didx[w][lane] = dst[e0 + lane]; }

  // stage 16x64 edge_attr tile as bf16 A-fragments (coalesced fp32 reads)
  unsigned ebase = (unsigned)e0 * EDGEF;
#pragma unroll 4
  for (int j = 0; j < 32; j++) {
    int t = j * 32 + lane;                 // 0..1023
    int r = t >> 6, k = t & 63;
    unsigned short b = f2bf(edge_attr[ebase + (unsigned)t]);
    afrag[w][(k >> 5) * 512 + a_lane(r, k & 31) * 16 + a_elem(k & 31)] = b;
  }
  __syncthreads();

  v16bf a0 = *(const v16bf*)&afrag[w][0 * 512 + lane * 16];
  v16bf a1 = *(const v16bf*)&afrag[w][1 * 512 + lane * 16];
  int n_lo = lane & 15, mhi = (lane >> 4) << 3;

  // hoist per-row u32 element offsets (invariant over the 8 nt-tiles)
  unsigned sbase[8], dbase[8];
#pragma unroll
  for (int r = 0; r < 8; r++) {
    sbase[r] = (unsigned)sidx[w][mhi + r] * HIDDEN;
    dbase[r] = (unsigned)didx[w][mhi + r] * HIDDEN;
  }

#pragma unroll
  for (int nt = 0; nt < 8; nt++) {
    v8f c = {};
    v16bf b0 = *(const v16bf*)&eWf[((0 * 8 + nt) * 32 + lane) * 16];
    v16bf b1 = *(const v16bf*)&eWf[((1 * 8 + nt) * 32 + lane) * 16];
    c = __builtin_amdgcn_wmma_f32_16x16x32_bf16(false, a0, false, b0, (short)0, c, false, false);
    c = __builtin_amdgcn_wmma_f32_16x16x32_bf16(false, a1, false, b1, (short)0, c, false, false);
    unsigned n = nt * 16 + n_lo;
    float bias = eb[n];
#pragma unroll
    for (int r = 0; r < 8; r++) {
      float v = fmaxf(c[r] + bias + x[sbase[r] + n], 0.0f);
      atomicAdd(&aggr[dbase[r] + n], v);
    }
  }
}

// ---------------------------------------------------------------------------
// GIN node MLP: x_out = relu(relu(h @ W1 + b1) @ W2 + b2), h = aggr (= x + sum)
// One wave per 16 nodes; hidden re-swizzled through LDS between the GEMMs.
// ---------------------------------------------------------------------------
__global__ __launch_bounds__(128) void gin_node_mlp(
    const float* __restrict__ hin, const unsigned short* __restrict__ W1f,
    const float* __restrict__ b1, const unsigned short* __restrict__ W2f,
    const float* __restrict__ b2, float* __restrict__ xout) {
  __shared__ __align__(32) unsigned short frag[4][4 * 512];  // 4 K-chunks
  int tid = threadIdx.x, w = tid >> 5, lane = tid & 31;
  int row0 = (blockIdx.x * 4 + w) * 16;

#pragma unroll 4
  for (int j = 0; j < 64; j++) {
    int t = j * 32 + lane;                 // 0..2047
    int r = t >> 7, k = t & 127;
    int rr = row0 + r; rr = rr < NNODES ? rr : NNODES - 1;
    unsigned short b = f2bf(hin[(unsigned)rr * HIDDEN + (unsigned)k]);
    frag[w][(k >> 5) * 512 + a_lane(r, k & 31) * 16 + a_elem(k & 31)] = b;
  }
  __syncthreads();

  v16bf a[4];
#pragma unroll
  for (int kc = 0; kc < 4; kc++) a[kc] = *(const v16bf*)&frag[w][kc * 512 + lane * 16];
  __syncthreads();

  int n_lo = lane & 15, mhi = (lane >> 4) << 3;
  // GEMM1 + ReLU, write hidden back into LDS in A-fragment layout
#pragma unroll
  for (int nt = 0; nt < 8; nt++) {
    v8f c = {};
#pragma unroll
    for (int kc = 0; kc < 4; kc++) {
      v16bf bf = *(const v16bf*)&W1f[((kc * 8 + nt) * 32 + lane) * 16];
      c = __builtin_amdgcn_wmma_f32_16x16x32_bf16(false, a[kc], false, bf, (short)0, c, false, false);
    }
    int n = nt * 16 + n_lo;
    float bias = b1[n];
#pragma unroll
    for (int r = 0; r < 8; r++) {
      int m = mhi + r;
      float v = fmaxf(c[r] + bias, 0.0f);
      frag[w][(n >> 5) * 512 + a_lane(m, n & 31) * 16 + a_elem(n & 31)] = f2bf(v);
    }
  }
  __syncthreads();

  v16bf h[4];
#pragma unroll
  for (int kc = 0; kc < 4; kc++) h[kc] = *(const v16bf*)&frag[w][kc * 512 + lane * 16];

  // per-row u32 output offsets
  unsigned obase[8];
  bool oin[8];
#pragma unroll
  for (int r = 0; r < 8; r++) {
    int rr = row0 + mhi + r;
    oin[r]   = rr < NNODES;
    obase[r] = (unsigned)(oin[r] ? rr : 0) * HIDDEN;
  }

  // GEMM2 + ReLU -> x_out
#pragma unroll
  for (int nt = 0; nt < 8; nt++) {
    v8f c = {};
#pragma unroll
    for (int kc = 0; kc < 4; kc++) {
      v16bf bf = *(const v16bf*)&W2f[((kc * 8 + nt) * 32 + lane) * 16];
      c = __builtin_amdgcn_wmma_f32_16x16x32_bf16(false, h[kc], false, bf, (short)0, c, false, false);
    }
    unsigned n = nt * 16 + n_lo;
    float bias = b2[n];
#pragma unroll
    for (int r = 0; r < 8; r++) {
      if (oin[r]) xout[obase[r] + n] = fmaxf(c[r] + bias, 0.0f);
    }
  }
}

// ---------------------------------------------------------------------------
// Edge predictor: logits = relu(concat(x[src],x[dst]) @ pW1 + pb1) @ pW2 + pb2
// One wave per 16 edges (K = 256 -> 8 K-chunks, 64 WMMAs per wave).
// ---------------------------------------------------------------------------
__global__ __launch_bounds__(128) void edge_predictor(
    const float* __restrict__ x, const int* __restrict__ src,
    const int* __restrict__ dst, const unsigned short* __restrict__ pW1f,
    const float* __restrict__ pb1, const float* __restrict__ pW2,
    const float* __restrict__ pb2, float* __restrict__ out) {
  __shared__ __align__(32) unsigned short frag[4][8 * 512];   // 8 KB / wave
  __shared__ int sidx[4][16], didx[4][16];
  __shared__ float part[4][32];
  int tid = threadIdx.x, w = tid >> 5, lane = tid & 31;
  int e0 = (blockIdx.x * 4 + w) * 16;

  if (lane < 16) { sidx[w][lane] = src[e0 + lane]; didx[w][lane] = dst[e0 + lane]; }
  __syncthreads();

  // gather 16 x 256 concat tile, convert to bf16 A-fragments (u32 offsets)
#pragma unroll 4
  for (int j = 0; j < 128; j++) {
    int t = j * 32 + lane;                 // 0..4095
    int r = t >> 8, k = t & 255;
    unsigned node = (unsigned)((k < HIDDEN) ? sidx[w][r] : didx[w][r]);
    float v = x[node * HIDDEN + (unsigned)(k & (HIDDEN - 1))];
    frag[w][(k >> 5) * 512 + a_lane(r, k & 31) * 16 + a_elem(k & 31)] = f2bf(v);
  }
  __syncthreads();

  v16bf a[8];
#pragma unroll
  for (int kc = 0; kc < 8; kc++) a[kc] = *(const v16bf*)&frag[w][kc * 512 + lane * 16];
  __syncthreads();

  float* hidden = (float*)&frag[w][0];     // reuse LDS: 16 x 128 fp32 = 8 KB
  int n_lo = lane & 15, mhi = (lane >> 4) << 3;
#pragma unroll
  for (int nt = 0; nt < 8; nt++) {
    v8f c = {};
#pragma unroll
    for (int kc = 0; kc < 8; kc++) {
      v16bf bf = *(const v16bf*)&pW1f[((kc * 8 + nt) * 32 + lane) * 16];
      c = __builtin_amdgcn_wmma_f32_16x16x32_bf16(false, a[kc], false, bf, (short)0, c, false, false);
    }
    int n = nt * 16 + n_lo;
    float bias = pb1[n];
#pragma unroll
    for (int r = 0; r < 8; r++) hidden[(mhi + r) * HIDDEN + n] = fmaxf(c[r] + bias, 0.0f);
  }
  __syncthreads();

  // final dot-128 with pW2: two lanes per edge (64 elems each), combine in LDS
  int m = lane & 15, half = lane >> 4;
  float acc = 0.0f;
#pragma unroll 8
  for (int k = half * 64; k < half * 64 + 64; k++) acc += hidden[m * HIDDEN + k] * pW2[k];
  part[w][lane] = acc;
  __syncthreads();
  if (lane < 16) out[e0 + lane] = part[w][lane] + part[w][lane + 16] + pb2[0];
}

// ---------------------------------------------------------------------------
extern "C" void kernel_launch(void* const* d_in, const int* in_sizes, int n_in,
                              void* d_out, int out_size, void* d_ws, size_t ws_size,
                              hipStream_t stream) {
  (void)in_sizes; (void)n_in; (void)out_size; (void)ws_size;
  const float* x_in  = (const float*)d_in[0];
  const int*   eidx  = (const int*)d_in[1];
  const float* eattr = (const float*)d_in[2];
  const float* edgeW = (const float*)d_in[3];
  const float* edgeb = (const float*)d_in[4];
  const float* W1    = (const float*)d_in[5];
  const float* b1    = (const float*)d_in[6];
  const float* W2    = (const float*)d_in[7];
  const float* b2    = (const float*)d_in[8];
  const float* pW1   = (const float*)d_in[9];
  const float* pb1   = (const float*)d_in[10];
  const float* pW2   = (const float*)d_in[11];
  const float* pb2   = (const float*)d_in[12];
  const int* src  = eidx;
  const int* dstp = eidx + NEDGES;

  // workspace layout: bf16 packed weights, then fp32 node buffers
  unsigned short* wf = (unsigned short*)d_ws;
  size_t off = 0;
  unsigned short* eWf  = wf + off; off += (size_t)NLAYER * 2 * 8 * 512;   // 3 x 8192
  unsigned short* W1f  = wf + off; off += (size_t)NLAYER * 4 * 8 * 512;   // 3 x 16384
  unsigned short* W2f  = wf + off; off += (size_t)NLAYER * 4 * 8 * 512;   // 3 x 16384
  unsigned short* pW1f = wf + off; off += (size_t)8 * 8 * 512;            // 32768
  size_t fbytes = (off * 2 + 255) & ~(size_t)255;
  float* xbuf = (float*)((char*)d_ws + fbytes);
  float* aggr = xbuf + (size_t)NNODES * HIDDEN;

  // one-time: pack all weights into bf16 B-fragment layout (L2-resident, 0.3 MB)
  for (int l = 0; l < NLAYER; l++) {
    pack_weights<<<32, 256, 0, stream>>>(edgeW + (size_t)l * EDGEF * HIDDEN, eWf + l * 8192, EDGEF, HIDDEN);
    pack_weights<<<64, 256, 0, stream>>>(W1 + (size_t)l * HIDDEN * HIDDEN, W1f + l * 16384, HIDDEN, HIDDEN);
    pack_weights<<<64, 256, 0, stream>>>(W2 + (size_t)l * HIDDEN * HIDDEN, W2f + l * 16384, HIDDEN, HIDDEN);
  }
  pack_weights<<<128, 256, 0, stream>>>(pW1, pW1f, 2 * HIDDEN, HIDDEN);

  hipMemcpyAsync(xbuf, x_in, (size_t)NNODES * HIDDEN * sizeof(float),
                 hipMemcpyDeviceToDevice, stream);

  for (int l = 0; l < NLAYER; l++) {
    // aggr = x  (so message atomics produce h = x + segment_sum)
    hipMemcpyAsync(aggr, xbuf, (size_t)NNODES * HIDDEN * sizeof(float),
                   hipMemcpyDeviceToDevice, stream);
    gine_message<<<NEDGES / 64, 128, 0, stream>>>(
        eattr, src, dstp, eWf + l * 8192, edgeb + (size_t)l * HIDDEN, xbuf, aggr);
    gin_node_mlp<<<(NNODES + 63) / 64, 128, 0, stream>>>(
        aggr, W1f + l * 16384, b1 + (size_t)l * HIDDEN,
        W2f + l * 16384, b2 + (size_t)l * HIDDEN, xbuf);
  }
  edge_predictor<<<NEDGES / 64, 128, 0, stream>>>(
      xbuf, src, dstp, pW1f, pb1, pW2, pb2, (float*)d_out);
}
